// GAT_43112881717637
// MI455X (gfx1250) — matrix operand order repaired and checked by reading, not compiled, not run on hardware
//
#include <hip/hip_runtime.h>
#include <hip/hip_bf16.h>
#include <math.h>

// ---------------------------------------------------------------------------
// GAT 2-layer pipeline for gfx1250 (MI455X), round 2.
// GEMMs: v_wmma_f32_16x16x32_bf16 with bf16x2 error-compensated split.
// A/B operands are PRE-PACKED into the exact WMMA per-lane register layout
// (hi & lo bf16 planes), so the GEMM inner loop is pure b128 loads + WMMA.
// Each wave computes a 16x64 C strip (4 N-tiles) to amortize A fetches.
// Edge softmax/aggregation: wave-per-edge, lane-contiguous atomics (L2-side).
// ---------------------------------------------------------------------------

typedef __bf16 bf16_t;
typedef bf16_t v16bf __attribute__((ext_vector_type(16)));
typedef float  v8f   __attribute__((ext_vector_type(8)));

#define NEG_SLOPE 0.2f
#define EPS_ATT   1e-16f

// monotone mapping float -> uint for atomicMax-based segment max
__device__ __forceinline__ unsigned encodeF(float x) {
    unsigned u = __float_as_uint(x);
    return u ^ (unsigned)((((int)u) >> 31) | 0x80000000);
}
__device__ __forceinline__ float decodeF(unsigned e) {
    unsigned u = e ^ ((e & 0x80000000u) ? 0x80000000u : 0xFFFFFFFFu);
    return __uint_as_float(u);
}

// ---------------------------- init kernels ---------------------------------
__global__ void fill_f32(float* p, long long n, float v) {
    long long i = (long long)blockIdx.x * blockDim.x + threadIdx.x;
    if (i < n) p[i] = v;
}
__global__ void fill_u32(unsigned* p, long long n, unsigned v) {
    long long i = (long long)blockIdx.x * blockDim.x + threadIdx.x;
    if (i < n) p[i] = v;
}

// --------------------- operand packing (WMMA layouts) ----------------------
// A layout (16-bit A, 16x32): per lane 16 bf16 contiguous.
//   lanes 0-15 : row = tileM*16+lane,   K = kb*32 + {0..7, 16..23}
//   lanes 16-31: row = tileM*16+lane-16,K = kb*32 + {8..15, 24..31}
// index = ((tileM*nKb + kb)*32 + lane)*16 + i
__global__ void pack_a_bf16(const float* __restrict__ A, bf16_t* __restrict__ hi,
                            bf16_t* __restrict__ lo, int M, int K)
{
    long long t = (long long)blockIdx.x * blockDim.x + threadIdx.x;
    const int nKb = K >> 5;
    const long long total = (long long)((M + 15) >> 4) * nKb * 32;
    if (t >= total) return;
    const int lane = (int)(t & 31);
    const long long tk = t >> 5;
    const int kb = (int)(tk % nKb);
    const int tileM = (int)(tk / nKb);
    const int half = lane >> 4, l16 = lane & 15;
    int row = tileM * 16 + l16; if (row >= M) row = M - 1;
    const float* src = A + (size_t)row * K + kb * 32 + (half ? 8 : 0);
    bf16_t* ph = hi + t * 16; bf16_t* pl = lo + t * 16;
#pragma unroll
    for (int i = 0; i < 8; ++i) {
        float f = src[i];      bf16_t h = (bf16_t)f;
        ph[i] = h;     pl[i] = (bf16_t)(f - (float)h);
    }
#pragma unroll
    for (int i = 0; i < 8; ++i) {
        float f = src[16 + i]; bf16_t h = (bf16_t)f;
        ph[8 + i] = h; pl[8 + i] = (bf16_t)(f - (float)h);
    }
}

// Fused layer-1 epilogue + layer-2 A pack: v = elu(acc[row,col] + b[col])
__global__ void pack_a_bias_elu_bf16(const float* __restrict__ A, const float* __restrict__ b,
                                     bf16_t* __restrict__ hi, bf16_t* __restrict__ lo,
                                     int M, int K)
{
    long long t = (long long)blockIdx.x * blockDim.x + threadIdx.x;
    const int nKb = K >> 5;
    const long long total = (long long)((M + 15) >> 4) * nKb * 32;
    if (t >= total) return;
    const int lane = (int)(t & 31);
    const long long tk = t >> 5;
    const int kb = (int)(tk % nKb);
    const int tileM = (int)(tk / nKb);
    const int half = lane >> 4, l16 = lane & 15;
    int row = tileM * 16 + l16; if (row >= M) row = M - 1;
    const int k0 = kb * 32 + (half ? 8 : 0);
    const float* src = A + (size_t)row * K + k0;
    bf16_t* ph = hi + t * 16; bf16_t* pl = lo + t * 16;
#pragma unroll
    for (int i = 0; i < 8; ++i) {
        float f = src[i] + b[k0 + i];
        f = (f > 0.f) ? f : (expf(f) - 1.f);
        bf16_t h = (bf16_t)f; ph[i] = h; pl[i] = (bf16_t)(f - (float)h);
    }
#pragma unroll
    for (int i = 0; i < 8; ++i) {
        float f = src[16 + i] + b[k0 + 16 + i];
        f = (f > 0.f) ? f : (expf(f) - 1.f);
        bf16_t h = (bf16_t)f; ph[8 + i] = h; pl[8 + i] = (bf16_t)(f - (float)h);
    }
}

// B layout (16-bit B, 32x16): per lane 16 bf16 contiguous.
//   col = tileN*16 + lane%16, K = kb*32 + (lane<16 ? 0 : 8) + (i<8 ? i : i+8)
// index = ((kb*tilesN + tileN)*32 + lane)*16 + i
__global__ void pack_b_bf16(const float* __restrict__ B, bf16_t* __restrict__ hi,
                            bf16_t* __restrict__ lo, int K, int Nout)
{
    long long t = (long long)blockIdx.x * blockDim.x + threadIdx.x;
    const int tilesN = Nout >> 4;
    const long long total = (long long)(K >> 5) * tilesN * 32;
    if (t >= total) return;
    const int lane = (int)(t & 31);
    const long long tk = t >> 5;
    const int tileN = (int)(tk % tilesN);
    const int kb = (int)(tk / tilesN);
    const int half = lane >> 4, l16 = lane & 15;
    const int col = tileN * 16 + l16;
    const float* src = B + (size_t)(kb * 32 + (half ? 8 : 0)) * Nout + col;
    bf16_t* ph = hi + t * 16; bf16_t* pl = lo + t * 16;
#pragma unroll
    for (int i = 0; i < 16; ++i) {
        float f = src[(size_t)(i + (i & 8)) * Nout];
        bf16_t h = (bf16_t)f; ph[i] = h; pl[i] = (bf16_t)(f - (float)h);
    }
}

// --------------------- WMMA GEMM on packed operands ------------------------
// C[M,Nout] = A*B. One wave computes a 16x64 strip (4 N-tiles); 8 waves/block.
// Requires K%32==0, Nout%64==0.
__global__ __launch_bounds__(256)
void gemm_wmma_packed(const bf16_t* __restrict__ Ah, const bf16_t* __restrict__ Al,
                      const bf16_t* __restrict__ Bh, const bf16_t* __restrict__ Bl,
                      float* __restrict__ C, int M, int K, int Nout)
{
    const int wave = threadIdx.x >> 5;
    const int lane = threadIdx.x & 31;
    const int nKb = K >> 5;
    const int tilesN = Nout >> 4;
    const int nGroups = tilesN >> 2;                 // 4 N-tiles per wave
    const int tilesM = (M + 15) >> 4;
    const int g = blockIdx.x * 8 + wave;
    if (g >= tilesM * nGroups) return;               // wave-uniform exit
    const int tileM = g / nGroups;
    const int tileN0 = (g % nGroups) * 4;
    const int half = lane >> 4, l16 = lane & 15;

    const v16bf* pAh = (const v16bf*)Ah;
    const v16bf* pAl = (const v16bf*)Al;
    const v16bf* pBh = (const v16bf*)Bh;
    const v16bf* pBl = (const v16bf*)Bl;

    v8f acc[4] = {};
    for (int kb = 0; kb < nKb; ++kb) {
        const size_t ai = (size_t)(tileM * nKb + kb) * 32 + lane;
        v16bf ah = pAh[ai];
        v16bf al = pAl[ai];
#pragma unroll
        for (int tnt = 0; tnt < 4; ++tnt) {
            const size_t bi = (size_t)(kb * tilesN + tileN0 + tnt) * 32 + lane;
            v16bf bh = pBh[bi];
            v16bf bl = pBl[bi];
            acc[tnt] = __builtin_amdgcn_wmma_f32_16x16x32_bf16(false, ah, false, bh,
                                                               (short)0, acc[tnt], false, false);
            acc[tnt] = __builtin_amdgcn_wmma_f32_16x16x32_bf16(false, ah, false, bl,
                                                               (short)0, acc[tnt], false, false);
            acc[tnt] = __builtin_amdgcn_wmma_f32_16x16x32_bf16(false, al, false, bh,
                                                               (short)0, acc[tnt], false, false);
        }
    }
    // C layout: VGPR r -> row (tileM*16 + half*8 + r), col = (tileN0+t)*16 + l16
    float* cb = C + (size_t)(tileM * 16 + half * 8) * Nout + l16;
    if (tileM * 16 + 16 <= M) {                      // full tile: straight stores
#pragma unroll
        for (int tnt = 0; tnt < 4; ++tnt)
#pragma unroll
            for (int r = 0; r < 8; ++r)
                cb[(size_t)r * Nout + (tileN0 + tnt) * 16] = acc[tnt][r];
    } else {
#pragma unroll
        for (int tnt = 0; tnt < 4; ++tnt)
#pragma unroll
            for (int r = 0; r < 8; ++r)
                if (tileM * 16 + half * 8 + r < M)
                    cb[(size_t)r * Nout + (tileN0 + tnt) * 16] = acc[tnt][r];
    }
}

// -------------------- attention coefficient projections --------------------
__global__ void alpha_proj(const float* __restrict__ xw,
                           const float* __restrict__ a_src,
                           const float* __restrict__ a_dst,
                           float* __restrict__ asrc, float* __restrict__ adst,
                           long long NH, int H, int Cc)
{
    long long i = (long long)blockIdx.x * blockDim.x + threadIdx.x;
    if (i >= NH) return;
    int h = (int)(i % H);
    const float* v  = xw + i * Cc;
    const float* as = a_src + (size_t)h * Cc;
    const float* ad = a_dst + (size_t)h * Cc;
    float s = 0.f, d = 0.f;
    for (int c = 0; c < Cc; ++c) { float x = v[c]; s += x * as[c]; d += x * ad[c]; }
    asrc[i] = s; adst[i] = d;
}

__device__ __forceinline__ void edge_sd(const int* __restrict__ ei, long long E,
                                        long long e, int& s, int& d)
{
    if (e < E) { s = ei[e]; d = ei[E + e]; }
    else       { s = d = (int)(e - E); }            // implicit self-loop
}

// pass A: segment max of leaky_relu(asrc[s]+adst[d]) per (dst, head)
__global__ void edge_max(const int* __restrict__ ei, long long E, long long ET,
                         const float* __restrict__ asrc, const float* __restrict__ adst,
                         unsigned* __restrict__ menc, int H)
{
    long long t = (long long)blockIdx.x * blockDim.x + threadIdx.x;
    if (t >= ET * H) return;
    long long e = t / H; int h = (int)(t % H);
    int s, d; edge_sd(ei, E, e, s, d);
    float v = asrc[(long long)s * H + h] + adst[(long long)d * H + h];
    v = (v > 0.f) ? v : NEG_SLOPE * v;
    atomicMax(&menc[(long long)d * H + h], encodeF(v));
}

// pass B: denom[d,h] += exp(e - m[d,h])
__global__ void edge_expsum(const int* __restrict__ ei, long long E, long long ET,
                            const float* __restrict__ asrc, const float* __restrict__ adst,
                            const unsigned* __restrict__ menc, float* __restrict__ denom, int H)
{
    long long t = (long long)blockIdx.x * blockDim.x + threadIdx.x;
    if (t >= ET * H) return;
    long long e = t / H; int h = (int)(t % H);
    int s, d; edge_sd(ei, E, e, s, d);
    float v = asrc[(long long)s * H + h] + adst[(long long)d * H + h];
    v = (v > 0.f) ? v : NEG_SLOPE * v;
    float m = decodeF(menc[(long long)d * H + h]);
    atomicAdd(&denom[(long long)d * H + h], expf(v - m));
}

// pass C: acc[d,h,:] += alpha * xw[s,h,:]  — one wave32 per edge,
// lane-contiguous channels so each atomic op covers a 128B burst.
__global__ __launch_bounds__(256)
void edge_aggregate_wave(const int* __restrict__ ei, long long E, long long ET,
                         const float* __restrict__ asrc, const float* __restrict__ adst,
                         const unsigned* __restrict__ menc, const float* __restrict__ denom,
                         const float* __restrict__ xw, float* __restrict__ acc,
                         int H, int Cc)
{
    long long w = ((long long)blockIdx.x * blockDim.x + threadIdx.x) >> 5;
    const int lane = threadIdx.x & 31;
    if (w >= ET) return;
    int s, d; edge_sd(ei, E, w, s, d);
    const float* xrow = xw  + (size_t)s * H * Cc;
    float*       orow = acc + (size_t)d * H * Cc;
    __builtin_prefetch(xrow + lane, 0, 1);          // global_prefetch_b8
    for (int h = 0; h < H; ++h) {
        float v = asrc[(size_t)s * H + h] + adst[(size_t)d * H + h];
        v = (v > 0.f) ? v : NEG_SLOPE * v;
        float m = decodeF(menc[(size_t)d * H + h]);
        float wgt = expf(v - m) / (denom[(size_t)d * H + h] + EPS_ATT);
        const float* xs = xrow + (size_t)h * Cc;
        float*       od = orow + (size_t)h * Cc;
        for (int c = lane; c < Cc; c += 32)
            atomicAdd(&od[c], wgt * xs[c]);
    }
}

// out = log_softmax(acc2 + b2) over 64 channels; one wave32 per node.
__global__ __launch_bounds__(256)
void bias_logsoftmax64(const float* __restrict__ acc2, const float* __restrict__ b2,
                       float* __restrict__ out, long long Nn)
{
    long long w = ((long long)blockIdx.x * blockDim.x + threadIdx.x) >> 5;
    int lane = threadIdx.x & 31;
    if (w >= Nn) return;
    const float* row = acc2 + w * 64;
    float v0 = row[lane]      + b2[lane];
    float v1 = row[lane + 32] + b2[lane + 32];
    float mx = fmaxf(v0, v1);
    for (int off = 16; off > 0; off >>= 1) mx = fmaxf(mx, __shfl_xor(mx, off, 32));
    float s = expf(v0 - mx) + expf(v1 - mx);
    for (int off = 16; off > 0; off >>= 1) s += __shfl_xor(s, off, 32);
    float lse = logf(s) + mx;
    out[w * 64 + lane]      = v0 - lse;
    out[w * 64 + lane + 32] = v1 - lse;
}

// ---------------------------------------------------------------------------
extern "C" void kernel_launch(void* const* d_in, const int* in_sizes, int n_in,
                              void* d_out, int out_size, void* d_ws, size_t ws_size,
                              hipStream_t stream)
{
    (void)n_in; (void)out_size; (void)ws_size;
    const float* x   = (const float*)d_in[0];
    const int*   ei  = (const int*)  d_in[1];   // [2,E] int32
    const float* W1  = (const float*)d_in[2];
    const float* as1 = (const float*)d_in[3];
    const float* ad1 = (const float*)d_in[4];
    const float* b1  = (const float*)d_in[5];
    const float* W2  = (const float*)d_in[6];
    const float* as2 = (const float*)d_in[7];
    const float* ad2 = (const float*)d_in[8];
    const float* b2  = (const float*)d_in[9];
    float* out = (float*)d_out;

    const int Fin = 512, H1 = 8, C1 = 32, F1 = 256, C2 = 64;  // per reference
    const long long N  = in_sizes[0] / Fin;    // 50000
    const long long E  = in_sizes[1] / 2;      // 800000
    const long long ET = E + N;                // + self loops
    const long long tilesM = (N + 15) / 16;

    // ---- workspace carve-out
    char* ws = (char*)d_ws; size_t off = 0;
    auto carve = [&](size_t bytes) { void* p = ws + off; off = (off + bytes + 255) & ~(size_t)255; return p; };
    float*    xw1   = (float*)   carve((size_t)N * F1 * 4);
    float*    acc1  = (float*)   carve((size_t)N * F1 * 4);
    float*    xw2   = (float*)   carve((size_t)N * C2 * 4);
    float*    acc2  = (float*)   carve((size_t)N * C2 * 4);
    bf16_t*   Ah1   = (bf16_t*)  carve((size_t)N * Fin * 2);  // packed x (reused for packed h)
    bf16_t*   Al1   = (bf16_t*)  carve((size_t)N * Fin * 2);
    bf16_t*   Bh1   = (bf16_t*)  carve((size_t)Fin * F1 * 2);
    bf16_t*   Bl1   = (bf16_t*)  carve((size_t)Fin * F1 * 2);
    bf16_t*   Bh2   = (bf16_t*)  carve((size_t)F1 * C2 * 2);
    bf16_t*   Bl2   = (bf16_t*)  carve((size_t)F1 * C2 * 2);
    float*    asrc1 = (float*)   carve((size_t)N * H1 * 4);
    float*    adst1 = (float*)   carve((size_t)N * H1 * 4);
    unsigned* m1    = (unsigned*)carve((size_t)N * H1 * 4);
    float*    den1  = (float*)   carve((size_t)N * H1 * 4);
    float*    asrc2 = (float*)   carve((size_t)N * 4);
    float*    adst2 = (float*)   carve((size_t)N * 4);
    unsigned* m2    = (unsigned*)carve((size_t)N * 4);
    float*    den2  = (float*)   carve((size_t)N * 4);
    bf16_t*   Ah2   = Ah1;   // alias: packed x dead after GEMM1
    bf16_t*   Al2   = Al1;

    const int TB = 256;
    auto blocks = [](long long n, int tb) { return (unsigned)((n + tb - 1) / tb); };
    const unsigned ENC_NEG_INF = 0x007FFFFFu;   // encodeF(-inf)

    // ---- init scratch
    fill_f32<<<blocks(N * F1, TB), TB, 0, stream>>>(acc1, N * F1, 0.f);
    fill_f32<<<blocks(N * C2, TB), TB, 0, stream>>>(acc2, N * C2, 0.f);
    fill_f32<<<blocks(N * H1, TB), TB, 0, stream>>>(den1, N * H1, 0.f);
    fill_f32<<<blocks(N, TB), TB, 0, stream>>>(den2, N, 0.f);
    fill_u32<<<blocks(N * H1, TB), TB, 0, stream>>>(m1, N * H1, ENC_NEG_INF);
    fill_u32<<<blocks(N, TB), TB, 0, stream>>>(m2, N, ENC_NEG_INF);

    // ================= layer 1 =================
    {   // pack x and W1, then xw1 = x @ W1  (M=N, K=512, Nout=256)
        long long aT = tilesM * (Fin / 32) * 32;
        pack_a_bf16<<<blocks(aT, TB), TB, 0, stream>>>(x, Ah1, Al1, (int)N, Fin);
        long long bT = (long long)(Fin / 32) * (F1 / 16) * 32;
        pack_b_bf16<<<blocks(bT, TB), TB, 0, stream>>>(W1, Bh1, Bl1, Fin, F1);
        long long waves = tilesM * (F1 / 64);
        gemm_wmma_packed<<<blocks(waves, 8), TB, 0, stream>>>(Ah1, Al1, Bh1, Bl1, xw1,
                                                              (int)N, Fin, F1);
    }
    alpha_proj<<<blocks(N * H1, TB), TB, 0, stream>>>(xw1, as1, ad1, asrc1, adst1, N * H1, H1, C1);
    edge_max   <<<blocks(ET * H1, TB), TB, 0, stream>>>(ei, E, ET, asrc1, adst1, m1, H1);
    edge_expsum<<<blocks(ET * H1, TB), TB, 0, stream>>>(ei, E, ET, asrc1, adst1, m1, den1, H1);
    edge_aggregate_wave<<<blocks(ET * 32, TB), TB, 0, stream>>>(ei, E, ET, asrc1, adst1, m1, den1,
                                                                xw1, acc1, H1, C1);

    // ================= layer 2 =================
    {   // fused bias+ELU+pack of h, pack W2, then xw2 = h @ W2 (K=256, Nout=64)
        long long aT = tilesM * (F1 / 32) * 32;
        pack_a_bias_elu_bf16<<<blocks(aT, TB), TB, 0, stream>>>(acc1, b1, Ah2, Al2, (int)N, F1);
        long long bT = (long long)(F1 / 32) * (C2 / 16) * 32;
        pack_b_bf16<<<blocks(bT, TB), TB, 0, stream>>>(W2, Bh2, Bl2, F1, C2);
        long long waves = tilesM * (C2 / 64);
        gemm_wmma_packed<<<blocks(waves, 8), TB, 0, stream>>>(Ah2, Al2, Bh2, Bl2, xw2,
                                                              (int)N, F1, C2);
    }
    alpha_proj<<<blocks(N, TB), TB, 0, stream>>>(xw2, as2, ad2, asrc2, adst2, N, 1, C2);
    edge_max   <<<blocks(ET, TB), TB, 0, stream>>>(ei, E, ET, asrc2, adst2, m2, 1);
    edge_expsum<<<blocks(ET, TB), TB, 0, stream>>>(ei, E, ET, asrc2, adst2, m2, den2, 1);
    edge_aggregate_wave<<<blocks(ET * 32, TB), TB, 0, stream>>>(ei, E, ET, asrc2, adst2, m2, den2,
                                                                xw2, acc2, 1, C2);
    // mean over 1 head == identity; + bias, log_softmax
    bias_logsoftmax64<<<blocks(N * 32, TB), TB, 0, stream>>>(acc2, b2, out, N);
}
